// Grid2d_50165218017425
// MI455X (gfx1250) — compile-verified, matrix operand order checked by reading, not run on hardware
//
#include <hip/hip_runtime.h>
#include <hip/hip_fp16.h>

typedef __attribute__((ext_vector_type(16))) _Float16 v16h;
typedef __attribute__((ext_vector_type(8)))  _Float16 v8h;
typedef __attribute__((ext_vector_type(8)))  float    v8f;

#define BB   2
#define NN   20
#define FF   128
#define CO   64
#define GRID 32
#define LL   1024
#define HH   8
#define WW   32
#define HWSZ 256
#define PP   5120   // N*H*W
#define NA   360

// ---------------------------------------------------------------------------
// 1) Fused per-angle conv term:
//    rayL[a,h,w] = Wf0 * conv3x3(rays[a], W1 ch1..3) + Wf1 * conv5x5(rays[a], W2 ch1..3)
// ---------------------------------------------------------------------------
__global__ void __launch_bounds__(256) k_rayL(const float* __restrict__ rays,
                                              const float* __restrict__ W1,
                                              const float* __restrict__ W2,
                                              const float* __restrict__ Wf,
                                              float* __restrict__ rayL) {
  int a = blockIdx.x;
  int t = threadIdx.x;
  int h = t >> 5, w = t & 31;
  const float* R = rays + (size_t)a * 3 * HWSZ;
  float acc3 = 0.f, acc5 = 0.f;
  for (int c = 0; c < 3; c++) {
    const float* rc = R + c * HWSZ;
    const float* w1 = W1 + (1 + c) * 9;
    #pragma unroll
    for (int dh = -1; dh <= 1; dh++)
      #pragma unroll
      for (int dw = -1; dw <= 1; dw++) {
        int hh = (h + dh + HH) % HH, w2i = (w + dw + WW) % WW;
        acc3 += w1[(dh + 1) * 3 + (dw + 1)] * rc[hh * WW + w2i];
      }
    const float* w2 = W2 + (1 + c) * 25;
    #pragma unroll
    for (int dh = -2; dh <= 2; dh++)
      #pragma unroll
      for (int dw = -2; dw <= 2; dw++) {
        int hh = (h + dh + HH) % HH, w2i = (w + dw + WW) % WW;
        acc5 += w2[(dh + 2) * 5 + (dw + 2)] * rc[hh * WW + w2i];
      }
  }
  rayL[a * HWSZ + t] = Wf[0] * acc3 + Wf[1] * acc5;
}

// ---------------------------------------------------------------------------
// 2a) pano stats: max/mean over F=128 channels, per (b,n,h,w)
// ---------------------------------------------------------------------------
__global__ void __launch_bounds__(256) k_pano_stats(const float* __restrict__ near_feats,
                                                    float* __restrict__ pmax,
                                                    float* __restrict__ pmean) {
  int bn = blockIdx.x;      // (b*NN + n)
  int t  = threadIdx.x;     // hw
  const float* base = near_feats + (size_t)bn * FF * HWSZ + t;
  float mx = -3.4e38f, sm = 0.f;
  for (int f = 0; f < FF; f++) {
    float v = base[(size_t)f * HWSZ];
    mx = fmaxf(mx, v);
    sm += v;
  }
  pmax[bn * HWSZ + t]  = mx;
  pmean[bn * HWSZ + t] = sm * (1.0f / FF);
}

// 2b) fused pano conv term over channels 4 (max) and 5 (mean)
__global__ void __launch_bounds__(256) k_panoL(const float* __restrict__ pmax,
                                               const float* __restrict__ pmean,
                                               const float* __restrict__ W1,
                                               const float* __restrict__ W2,
                                               const float* __restrict__ Wf,
                                               float* __restrict__ panoL) {
  int bn = blockIdx.x;
  int t  = threadIdx.x;
  int h = t >> 5, w = t & 31;
  const float* pm = pmax  + bn * HWSZ;
  const float* pe = pmean + bn * HWSZ;
  float acc3 = 0.f, acc5 = 0.f;
  const float* w14 = W1 + 4 * 9;
  const float* w15 = W1 + 5 * 9;
  #pragma unroll
  for (int dh = -1; dh <= 1; dh++)
    #pragma unroll
    for (int dw = -1; dw <= 1; dw++) {
      int o = ((h + dh + HH) % HH) * WW + ((w + dw + WW) % WW);
      int k = (dh + 1) * 3 + (dw + 1);
      acc3 += w14[k] * pm[o] + w15[k] * pe[o];
    }
  const float* w24 = W2 + 4 * 25;
  const float* w25 = W2 + 5 * 25;
  #pragma unroll
  for (int dh = -2; dh <= 2; dh++)
    #pragma unroll
    for (int dw = -2; dw <= 2; dw++) {
      int o = ((h + dh + HH) % HH) * WW + ((w + dw + WW) % WW);
      int k = (dh + 2) * 5 + (dw + 2);
      acc5 += w24[k] * pm[o] + w25[k] * pe[o];
    }
  panoL[bn * HWSZ + t] = Wf[0] * acc3 + Wf[1] * acc5;
}

// ---------------------------------------------------------------------------
// 3) overhead stats: max/mean over Co=64 per grid cell
// ---------------------------------------------------------------------------
__global__ void __launch_bounds__(256) k_ovStats(const float* __restrict__ overhead,
                                                 float* __restrict__ ovStats) {
  int b = blockIdx.x;
  int t = threadIdx.x;
  for (int c = 0; c < 4; c++) {
    int s = c * 256 + t;
    float mx = -3.4e38f, sm = 0.f;
    for (int co = 0; co < CO; co++) {
      float v = overhead[((size_t)b * CO + co) * LL + s];
      mx = fmaxf(mx, v);
      sm += v;
    }
    ovStats[((size_t)b * LL + s) * 2 + 0] = mx;
    ovStats[((size_t)b * LL + s) * 2 + 1] = sm * (1.0f / CO);
  }
}

// ---------------------------------------------------------------------------
// 4) near_feats [b,n,f,hw] -> featsT f16 [b][f][p = n*256+hw]
//    (so WMMA B-fragment K runs are contiguous)
// ---------------------------------------------------------------------------
__global__ void __launch_bounds__(256) k_featsT(const float* __restrict__ near_feats,
                                                _Float16* __restrict__ featsT) {
  int idx = blockIdx.x;             // (b*NN + n)*FF + f
  int t   = threadIdx.x;            // hw
  int b = idx / (NN * FF);
  int r = idx % (NN * FF);
  int n = r / FF;
  int f = r % FF;
  float v = near_feats[(size_t)idx * HWSZ + t];
  featsT[((size_t)(b * FF + f)) * PP + n * HWSZ + t] = (_Float16)v;
}

// ---------------------------------------------------------------------------
// 5) Main: distance/bearing, logits, softmax over 5120, write weight (f32)
// ---------------------------------------------------------------------------
__global__ void __launch_bounds__(256) k_main(const float* __restrict__ bbox,
                                              const float* __restrict__ near_locs,
                                              const float* __restrict__ W1,
                                              const float* __restrict__ b1,
                                              const float* __restrict__ W2,
                                              const float* __restrict__ b2,
                                              const float* __restrict__ Wf,
                                              const float* __restrict__ bf,
                                              const float* __restrict__ rayL,
                                              const float* __restrict__ panoL,
                                              const float* __restrict__ ovStats,
                                              float* __restrict__ weight_out) {
  __shared__ float sD[NN];
  __shared__ int   sIdx[NN];
  __shared__ float sC[4];                        // cD, cM, cE, cB
  __shared__ __align__(16) float sLog[PP];
  __shared__ float sRed[8];

  int bs = blockIdx.x;
  int b  = bs >> 10;
  int s  = bs & 1023;
  int t  = threadIdx.x;

  if (t < NN) {
    int iy = s >> 5, ix = s & 31;
    float y0 = bbox[b * 4 + 0], x0 = bbox[b * 4 + 1];
    float y1 = bbox[b * 4 + 2], x1 = bbox[b * 4 + 3];
    float cy = y0 + (y1 - y0) * (float)iy * (1.0f / (GRID - 1));
    float cx = x0 + (x1 - x0) * (float)ix * (1.0f / (GRID - 1));
    float dy = near_locs[(b * NN + t) * 2 + 0] - cy;
    float dx = near_locs[(b * NN + t) * 2 + 1] - cx;
    sD[t] = sqrtf(dy * dy + dx * dx);
    float th = atan2f(dx, dy);                      // jnp.arctan2(diff_x, diff_y)
    if (th < 0.f) th += 6.283185307179586f;
    int ai = __float2int_rn(th * 57.29577951308232f);  // round-half-even like jnp.round
    ai %= NA;
    sIdx[t] = ai;
  }
  if (t == 0) {
    float s10 = 0.f, s16 = 0.f, s17 = 0.f, s20 = 0.f, s26 = 0.f, s27 = 0.f;
    for (int k = 0; k < 9; k++)  { s10 += W1[0 * 9 + k];  s16 += W1[6 * 9 + k];  s17 += W1[7 * 9 + k]; }
    for (int k = 0; k < 25; k++) { s20 += W2[0 * 25 + k]; s26 += W2[6 * 25 + k]; s27 += W2[7 * 25 + k]; }
    float wf0 = Wf[0], wf1 = Wf[1];
    sC[0] = wf0 * s10 + wf1 * s20;                      // distance coeff
    sC[1] = wf0 * s16 + wf1 * s26;                      // ov-max coeff
    sC[2] = wf0 * s17 + wf1 * s27;                      // ov-mean coeff
    sC[3] = wf0 * b1[0] + wf1 * b2[0] + bf[0];          // bias
  }
  __syncthreads();

  float ovm  = ovStats[((size_t)b * LL + s) * 2 + 0];
  float ove  = ovStats[((size_t)b * LL + s) * 2 + 1];
  float base = sC[1] * ovm + sC[2] * ove + sC[3];
  float cD   = sC[0];

  float lmax = -3.4e38f;
  for (int p = t; p < PP; p += 256) {
    int n = p >> 8, hw = p & 255;
    float v = cD * sD[n] + rayL[sIdx[n] * HWSZ + hw] + panoL[(b * NN + n) * HWSZ + hw] + base;
    sLog[p] = v;
    lmax = fmaxf(lmax, v);
  }
  // wave32 reduction, then cross-wave via LDS (8 waves)
  #pragma unroll
  for (int off = 16; off; off >>= 1) lmax = fmaxf(lmax, __shfl_xor(lmax, off, 32));
  if ((t & 31) == 0) sRed[t >> 5] = lmax;
  __syncthreads();
  float blkmax = sRed[0];
  #pragma unroll
  for (int i = 1; i < 8; i++) blkmax = fmaxf(blkmax, sRed[i]);

  float lsum = 0.f;
  for (int p = t; p < PP; p += 256) {
    float e = __expf(sLog[p] - blkmax);
    sLog[p] = e;
    lsum += e;
  }
  #pragma unroll
  for (int off = 16; off; off >>= 1) lsum += __shfl_xor(lsum, off, 32);
  __syncthreads();                   // protect sRed reuse
  if ((t & 31) == 0) sRed[t >> 5] = lsum;
  __syncthreads();
  float tot = 0.f;
  #pragma unroll
  for (int i = 0; i < 8; i++) tot += sRed[i];
  float inv = 1.0f / tot;

  // 42 MB stream: write as float4 (global_store_b128)
  float* wout = weight_out + (size_t)(b * LL + s) * PP;
  for (int p4 = t * 4; p4 < PP; p4 += 1024) {
    float4 v4;
    v4.x = sLog[p4 + 0] * inv;
    v4.y = sLog[p4 + 1] * inv;
    v4.z = sLog[p4 + 2] * inv;
    v4.w = sLog[p4 + 3] * inv;
    *(float4*)(wout + p4) = v4;
  }
}

// ---------------------------------------------------------------------------
// 6) WMMA GEMM: grid_feats[b, s, f] = sum_p weight[b,s,p] * featsT[b,f,p]
//    Per block: 16 s-rows x 128 f-cols; 8 waves, each a 16x16 tile.
//    K step 64 (2 WMMAs/iter), ping-pong A tile in LDS (1 barrier/iter).
//    Per-lane f16 fragment = two contiguous 16B runs -> b128 loads.
// ---------------------------------------------------------------------------
__global__ void __launch_bounds__(256) k_gemm(const float* __restrict__ weight,
                                              const _Float16* __restrict__ featsT,
                                              float* __restrict__ gout) {
  __shared__ __align__(16) float Atile[2][16 * 64];

  int blk   = blockIdx.x;
  int b     = blk >> 6;
  int sBase = (blk & 63) * 16;
  int t     = threadIdx.x;
  int wave  = t >> 5;
  int lane  = t & 31;
  int fBase = wave * 16;
  int m     = lane & 15;         // A row (M) / B column (N)
  int g     = lane >> 4;         // lane half-group

  const float*    Wt   = weight + ((size_t)b * LL + sBase) * PP;
  const _Float16* colp = featsT + ((size_t)b * FF + (fBase + m)) * PP;

  // cooperative A tile load: 16 rows x 64 K f32 = 1024 floats, one float4/thread
  int li   = t * 4;
  int lrow = li >> 6;
  int lcol = li & 63;
  const float* lsrc = Wt + (size_t)lrow * PP + lcol;

  *(float4*)(&Atile[0][li]) = *(const float4*)(lsrc);   // preload tile 0

  v8f acc = {};
  for (int kk = 0; kk < PP; kk += 64) {
    int cur = (kk >> 6) & 1;
    __syncthreads();                                    // tile[cur] ready
    if (kk + 64 < PP)                                   // prefetch next tile
      *(float4*)(&Atile[1 ^ cur][li]) = *(const float4*)(lsrc + kk + 64);
    if (kk + 128 < PP)                                  // keep B stream ahead (global_prefetch_b8)
      __builtin_prefetch(colp + kk + 128, 0, 1);

    const float* Ab = &Atile[cur][m * 64];
    #pragma unroll
    for (int ks = 0; ks < 64; ks += 32) {
      // half j -> K = ks + 8g + j (j<8) and ks + 16 + 8g + (j-8): two contiguous runs
      v16h a, bbv;
      const float* ap = Ab + ks + 8 * g;
      #pragma unroll
      for (int j = 0; j < 8; j++) {
        a[j]     = (_Float16)ap[j];
        a[j + 8] = (_Float16)ap[16 + j];
      }
      v8h b0 = *(const v8h*)(colp + kk + ks + 8 * g);        // global_load_b128
      v8h b1 = *(const v8h*)(colp + kk + ks + 16 + 8 * g);   // global_load_b128
      #pragma unroll
      for (int j = 0; j < 8; j++) {
        bbv[j]     = b0[j];
        bbv[j + 8] = b1[j];
      }
      acc = __builtin_amdgcn_wmma_f32_16x16x32_f16(
          /*neg_a=*/false, a, /*neg_b=*/false, bbv,
          /*c_mod=*/(short)0, acc, /*reuse_a=*/false, /*reuse_b=*/false);
    }
  }

  // C/D layout: VGPR r -> (M = r + 8*g, N = lane&15)
  int col = lane & 15;
  int rowOff = g * 8;
  #pragma unroll
  for (int r = 0; r < 8; r++) {
    gout[((size_t)(b * LL + sBase + rowOff + r)) * FF + fBase + col] = acc[r];
  }
}

// ---------------------------------------------------------------------------
extern "C" void kernel_launch(void* const* d_in, const int* in_sizes, int n_in,
                              void* d_out, int out_size, void* d_ws, size_t ws_size,
                              hipStream_t stream) {
  const float* bbox       = (const float*)d_in[0];
  const float* near_locs  = (const float*)d_in[1];
  const float* near_feats = (const float*)d_in[2];
  const float* overhead   = (const float*)d_in[3];
  const float* rays       = (const float*)d_in[4];
  const float* W1 = (const float*)d_in[5];
  const float* b1 = (const float*)d_in[6];
  const float* W2 = (const float*)d_in[7];
  const float* b2 = (const float*)d_in[8];
  const float* Wf = (const float*)d_in[9];
  const float* bf = (const float*)d_in[10];

  char* ws = (char*)d_ws;
  float*    rayL    = (float*)(ws + 0);        // 360*256*4      = 368640
  float*    pmax    = (float*)(ws + 368640);   // 2*20*256*4     =  40960
  float*    pmean   = (float*)(ws + 409600);   //                   40960
  float*    panoL   = (float*)(ws + 450560);   //                   40960
  float*    ovStats = (float*)(ws + 491520);   // 2*1024*2*4     =  16384
  _Float16* featsT  = (_Float16*)(ws + 507904);// 2*128*5120*2   = 2621440  -> ~3.13 MB total

  float* gout = (float*)d_out;                          // [B,32,32,128]
  float* wout = (float*)d_out + (size_t)BB * LL * FF;   // [B,32,32,20,8,32]

  k_rayL      <<<NA,            256, 0, stream>>>(rays, W1, W2, Wf, rayL);
  k_pano_stats<<<BB * NN,       256, 0, stream>>>(near_feats, pmax, pmean);
  k_panoL     <<<BB * NN,       256, 0, stream>>>(pmax, pmean, W1, W2, Wf, panoL);
  k_ovStats   <<<BB,            256, 0, stream>>>(overhead, ovStats);
  k_featsT    <<<BB * NN * FF,  256, 0, stream>>>(near_feats, featsT);
  k_main      <<<BB * LL,       256, 0, stream>>>(bbox, near_locs, W1, b1, W2, b2, Wf, bf,
                                                  rayL, panoL, ovStats, wout);
  k_gemm      <<<BB * (LL / 16), 256, 0, stream>>>(wout, featsT, gout);
}
